// GNNEncoder_721554506528
// MI455X (gfx1250) — matrix operand order repaired and checked by reading, not compiled, not run on hardware
//
#include <hip/hip_runtime.h>
#include <hip/hip_bf16.h>
#include <stdint.h>

#define NN   60000      // nodes
#define NE   600000     // edges
#define NG   512        // graphs
#define NF   128        // input feature dim
#define D_G1 256
#define D_G2 320
#define D_G21 448
#define D_E1 256
#define D_F1 256
#define MAXD 448

typedef __attribute__((ext_vector_type(16))) __bf16 v16bf;
typedef __attribute__((ext_vector_type(8)))  __bf16 v8bf;
typedef __attribute__((ext_vector_type(8)))  float  v8f;
typedef unsigned short u16;

__device__ __forceinline__ u16 f2bf(float f) {
    unsigned int u = __float_as_uint(f);
    u += 0x7FFFu + ((u >> 16) & 1u);          // round to nearest even
    return (u16)(u >> 16);
}

__device__ __forceinline__ v16bf pack16(v8bf lo, v8bf hi) {
    v16bf a;
#pragma unroll
    for (int i = 0; i < 8; ++i) { a[i] = lo[i]; a[i + 8] = hi[i]; }
    return a;
}

// ---------------------------------------------------------------- utilities
__global__ void k_fill_f32(float* p, long n, float v) {
    long t = (long)blockIdx.x * blockDim.x + threadIdx.x;
    if (t < n) p[t] = v;
}

__global__ void k_fill_i32(int* p, long n, int v) {
    long t = (long)blockIdx.x * blockDim.x + threadIdx.x;
    if (t < n) p[t] = v;
}

__global__ void k_copy_i32(const int* __restrict__ in, int* __restrict__ out, long n) {
    long t = (long)blockIdx.x * blockDim.x + threadIdx.x;
    if (t < n) out[t] = in[t];
}

__global__ void k_cvt_bf16(const float* __restrict__ in, u16* __restrict__ out, long n) {
    long t = (long)blockIdx.x * blockDim.x + threadIdx.x;
    if (t < n) out[t] = f2bf(in[t]);
}

// Wt[o][i] = bf16(W[i][o])
__global__ void k_transpose_bf16(const float* __restrict__ W, u16* __restrict__ Wt,
                                 int in_dim, int out_dim) {
    int t = blockIdx.x * blockDim.x + threadIdx.x;
    if (t < in_dim * out_dim) {
        int i = t / out_dim, o = t % out_dim;
        Wt[(size_t)o * in_dim + i] = f2bf(W[(size_t)i * out_dim + o]);
    }
}

// ------------------------------------------------------- CSR construction
__global__ void k_hist(const int* __restrict__ dst, int* __restrict__ cnt, int e) {
    int t = blockIdx.x * blockDim.x + threadIdx.x;
    if (t < e) atomicAdd(&cnt[dst[t]], 1);
}

__global__ void k_dinv_from_cnt(const int* __restrict__ cnt, float* __restrict__ dinv, int n) {
    int t = blockIdx.x * blockDim.x + threadIdx.x;
    if (t < n) dinv[t] = rsqrtf((float)cnt[t] + 1.0f);   // +1 = self loop
}

// single-block exclusive scan: rowptr[i] = sum_{j<i} cnt[j], rowptr[n] = total
__global__ __launch_bounds__(256) void k_scan_rowptr(const int* __restrict__ cnt,
                                                     int* __restrict__ rowptr, int n) {
    __shared__ int part[256];
    const int tid = threadIdx.x;
    const int chunk = (n + 255) / 256;
    const int lo = tid * chunk;
    const int hi = (lo + chunk < n) ? lo + chunk : n;
    int s = 0;
    for (int i = lo; i < hi; ++i) s += cnt[i];
    part[tid] = s;
    __syncthreads();
    if (tid == 0) {
        int run = 0;
        for (int i = 0; i < 256; ++i) { int t = part[i]; part[i] = run; run += t; }
    }
    __syncthreads();
    int run = part[tid];
    for (int i = lo; i < hi; ++i) { rowptr[i] = run; run += cnt[i]; }
    if (tid == 255) rowptr[n] = run;
}

__global__ void k_fill_eidx(const int* __restrict__ dst, int* __restrict__ cursor,
                            int* __restrict__ eidx, int e) {
    int t = blockIdx.x * blockDim.x + threadIdx.x;
    if (t < e) {
        int pos = atomicAdd(&cursor[dst[t]], 1);
        eidx[pos] = t;
    }
}

// ------------------------------------------------------------ WMMA GEMM
// out[rows,Nout] = act( Xbf[rows,K] @ Wt[Nout,K]^T + bias )
// one 16x16 tile per wave, 8 waves/block stacked in M (128 rows/block)
__global__ __launch_bounds__(256) void k_gemm_wmma(
    const u16* __restrict__ Xbf, const u16* __restrict__ Wt,
    const float* __restrict__ bias, float* __restrict__ outF,
    int rows, int K, int Nout, int act) {
    const int lane = threadIdx.x & 31;
    const int wv   = threadIdx.x >> 5;
    const int half = lane >> 4;
    const int l16  = lane & 15;
    const int tileN = blockIdx.x * 16;
    const int tileM = blockIdx.y * 128 + wv * 16;

    int row = tileM + l16;
    if (row >= rows) row = rows - 1;                 // clamp loads; stores guarded
    const int col = tileN + l16;

    const u16* pa = Xbf + (size_t)row * K + half * 8;   // A: M=lane%16, K split by half
    const u16* pb = Wt  + (size_t)col * K + half * 8;   // B: N=lane%16, K split by half

    v8f acc = {};
    for (int k0 = 0; k0 < K; k0 += 32) {
        __builtin_prefetch(pa + k0 + 64, 0, 0);
        v8bf a0 = *(const v8bf*)(pa + k0);
        v8bf a1 = *(const v8bf*)(pa + k0 + 16);
        v16bf a = pack16(a0, a1);
        v16bf b = pack16(*(const v8bf*)(pb + k0), *(const v8bf*)(pb + k0 + 16));
        acc = __builtin_amdgcn_wmma_f32_16x16x32_bf16(false, a, false, b,
                                                      (short)0, acc, false, false);
    }

    const float bb = bias ? bias[col] : 0.0f;
#pragma unroll
    for (int r = 0; r < 8; ++r) {
        int m = tileM + half * 8 + r;                // C: M = r + 8*half, N = lane%16
        if (m < rows) {
            float v = acc[r] + bb;
            if (act == 1) v = fmaxf(v, 0.0f);
            outF[(size_t)m * Nout + col] = v;
        }
    }
}

// ------------------------------------- fused GCN aggregate + bias + activation
// One wave per node; lanes stripe the feature dim (dim % 32 == 0, dim <= 448).
// out[i] = act( dinv[i]^2 * h[i] + sum_{e: dst[e]==i} dinv[src]*dinv[i]*h[src] + b )
// act: 2 = sigmoid, 3 = selu. Emits fp32 (pooling) and bf16 (next GEMM input).
__global__ __launch_bounds__(256) void k_aggregate_act(
    const float* __restrict__ h, const float* __restrict__ dinv,
    const int* __restrict__ rowptr, const int* __restrict__ eidx,
    const int* __restrict__ src, const float* __restrict__ bias,
    float* __restrict__ outF, u16* __restrict__ outBf,
    int nnodes, int dim, int act) {
    const int lane = threadIdx.x & 31;
    const int wv   = threadIdx.x >> 5;
    const int node = blockIdx.x * 8 + wv;
    if (node >= nnodes) return;

    const int nj = dim >> 5;                 // <= 14
    const float di = dinv[node];
    float acc[MAXD / 32];

    const float* hi = h + (size_t)node * dim;
    for (int j = 0; j < nj; ++j) acc[j] = hi[lane + (j << 5)] * di * di;

    const int p1 = rowptr[node + 1];
    for (int p = rowptr[node]; p < p1; ++p) {
        const int s = src[eidx[p]];
        const float c = dinv[s] * di;
        const float* hs = h + (size_t)s * dim;
        for (int j = 0; j < nj; ++j) acc[j] += hs[lane + (j << 5)] * c;
    }

    const float al = 1.6732632423543772f, sc = 1.0507009873554805f;
    for (int j = 0; j < nj; ++j) {
        const int f = lane + (j << 5);
        float v = acc[j] + bias[f];
        if (act == 2) v = 1.0f / (1.0f + expf(-v));
        else          v = (v > 0.0f) ? sc * v : sc * al * (expf(v) - 1.0f);
        outF[(size_t)node * dim + f]  = v;
        outBf[(size_t)node * dim + f] = f2bf(v);
    }
}

// ------------------------------------------------- fused EdgeConv (16 edges/block)
// h = relu(cat[x_d, x_s - x_d] @ W1 + b1); o = h @ W2 + b2
// relu(segment_max(o)) == int-atomicMax of max(o,0) into zero-initialized buffer
__global__ __launch_bounds__(256) void k_edgeconv(
    const float* __restrict__ x, const int* __restrict__ src, const int* __restrict__ dst,
    const u16* __restrict__ W1t, const float* __restrict__ b1,
    const u16* __restrict__ W2t, const float* __restrict__ b2,
    float* __restrict__ out2) {
    __shared__ u16 Abuf[16 * 256];   // 16 edges x (2*NF) bf16
    __shared__ u16 Hbuf[16 * 256];   // 16 edges x E1 bf16 (post-ReLU hidden)
    __shared__ int sdst[16];

    const int  tid   = threadIdx.x;
    const long ebase = (long)blockIdx.x * 16;

    for (int idx = tid; idx < 16 * NF; idx += 256) {
        int e = idx >> 7, f = idx & (NF - 1);
        long ge = ebase + e; if (ge >= NE) ge = NE - 1;
        int d = dst[ge], s = src[ge];
        float xi = x[(size_t)d * NF + f];
        float xj = x[(size_t)s * NF + f];
        Abuf[e * 256 + f]      = f2bf(xi);
        Abuf[e * 256 + NF + f] = f2bf(xj - xi);
    }
    if (tid < 16) {
        long ge = ebase + tid; if (ge >= NE) ge = NE - 1;
        sdst[tid] = dst[ge];
    }
    __syncthreads();

    const int lane = tid & 31, wv = tid >> 5;
    const int half = lane >> 4, l16 = lane & 15;

    // ---- hidden = relu(A @ W1 + b1) : 16 x 256
#pragma unroll
    for (int t = 0; t < 2; ++t) {
        const int col = (wv * 2 + t) * 16 + l16;
        const u16* pb = W1t + (size_t)col * 256 + half * 8;
        const u16* pa = &Abuf[l16 * 256 + half * 8];
        v8f acc = {};
#pragma unroll
        for (int k0 = 0; k0 < 256; k0 += 32) {
            v16bf a = pack16(*(const v8bf*)(pa + k0), *(const v8bf*)(pa + k0 + 16));
            v16bf b = pack16(*(const v8bf*)(pb + k0), *(const v8bf*)(pb + k0 + 16));
            acc = __builtin_amdgcn_wmma_f32_16x16x32_bf16(false, a, false, b,
                                                          (short)0, acc, false, false);
        }
        const float bb = b1[col];
#pragma unroll
        for (int r = 0; r < 8; ++r) {
            int m = half * 8 + r;
            Hbuf[m * 256 + col] = f2bf(fmaxf(acc[r] + bb, 0.0f));
        }
    }
    __syncthreads();

    // ---- out = hidden @ W2 + b2, clamp at 0, atomicMax scatter
#pragma unroll
    for (int t = 0; t < 2; ++t) {
        const int col = (wv * 2 + t) * 16 + l16;
        const u16* pb = W2t + (size_t)col * 256 + half * 8;
        const u16* pa = &Hbuf[l16 * 256 + half * 8];
        v8f acc = {};
#pragma unroll
        for (int k0 = 0; k0 < 256; k0 += 32) {
            v16bf a = pack16(*(const v8bf*)(pa + k0), *(const v8bf*)(pa + k0 + 16));
            v16bf b = pack16(*(const v8bf*)(pb + k0), *(const v8bf*)(pb + k0 + 16));
            acc = __builtin_amdgcn_wmma_f32_16x16x32_bf16(false, a, false, b,
                                                          (short)0, acc, false, false);
        }
        const float bb = b2[col];
#pragma unroll
        for (int r = 0; r < 8; ++r) {
            int  m  = half * 8 + r;
            long ge = ebase + m;
            float v = fmaxf(acc[r] + bb, 0.0f);
            if (ge < NE && v > 0.0f)     // non-negative floats: int order == float order
                atomicMax((int*)&out2[(size_t)sdst[m] * D_F1 + col], __float_as_int(v));
        }
    }
}

// ---------------------------------------------------------------- pooling
__global__ void k_pool(const float* __restrict__ x1, const float* __restrict__ x2,
                       const int* __restrict__ batch, float* __restrict__ pooled, long total) {
    long t = (long)blockIdx.x * blockDim.x + threadIdx.x;
    if (t < total) {                                   // total = NN * 512
        long i = t >> 9;
        int  c = (int)(t & 511);
        float v = (c < 256) ? x1[i * 256 + c] : x2[i * 256 + (c - 256)];
        atomicAdd(&pooled[(size_t)batch[i] * 512 + c], v);
    }
}

// ================================================================= host
static inline unsigned ceil_div(long a, long b) { return (unsigned)((a + b - 1) / b); }

extern "C" void kernel_launch(void* const* d_in, const int* in_sizes, int n_in,
                              void* d_out, int out_size, void* d_ws, size_t ws_size,
                              hipStream_t stream) {
    (void)in_sizes; (void)n_in; (void)out_size; (void)ws_size;

    const float* x     = (const float*)d_in[0];
    const int*   ei    = (const int*)d_in[1];
    const int*   src   = ei;
    const int*   dst   = ei + NE;
    const int*   batch = (const int*)d_in[2];
    const float* W_g1  = (const float*)d_in[3];  const float* b_g1  = (const float*)d_in[4];
    const float* W_g2  = (const float*)d_in[5];  const float* b_g2  = (const float*)d_in[6];
    const float* W_g21 = (const float*)d_in[7];  const float* b_g21 = (const float*)d_in[8];
    const float* W_g32 = (const float*)d_in[9];  const float* b_g32 = (const float*)d_in[10];
    const float* W_e1  = (const float*)d_in[11]; const float* b_e1  = (const float*)d_in[12];
    const float* W_e2  = (const float*)d_in[13]; const float* b_e2  = (const float*)d_in[14];
    const float* W_fc1 = (const float*)d_in[15]; const float* b_fc1 = (const float*)d_in[16];
    float* out = (float*)d_out;

    // ---- workspace carve
    char* ws = (char*)d_ws; size_t off = 0;
    auto carve = [&](size_t bytes) -> void* {
        size_t o = (off + 255) & ~(size_t)255; off = o + bytes; return (void*)(ws + o);
    };
    float* dinv    = (float*)carve((size_t)NN * 4);
    int* cnt       = (int*)carve((size_t)NN * 4);
    int* rowptr    = (int*)carve((size_t)(NN + 1) * 4);
    int* cursor    = (int*)carve((size_t)NN * 4);
    int* eidx      = (int*)carve((size_t)NE * 4);
    u16* wt_g1     = (u16*)carve((size_t)NF   * D_G1  * 2);
    u16* wt_g2     = (u16*)carve((size_t)D_G1 * D_G2  * 2);
    u16* wt_g21    = (u16*)carve((size_t)D_G2 * D_G21 * 2);
    u16* wt_g32    = (u16*)carve((size_t)D_G21* D_E1  * 2);
    u16* wt_e1     = (u16*)carve((size_t)2*NF * D_E1  * 2);
    u16* wt_e2     = (u16*)carve((size_t)D_E1 * D_F1  * 2);
    u16* wt_fc1    = (u16*)carve((size_t)(D_E1+D_F1) * D_F1 * 2);
    u16* xbf       = (u16*)carve((size_t)NN * MAXD * 2);
    float* hbuf    = (float*)carve((size_t)NN * MAXD * 4);   // pre-activation h
    float* actf    = (float*)carve((size_t)NN * MAXD * 4);   // post-activation (fp32)
    float* x2      = (float*)carve((size_t)NN * D_F1 * 4);
    float* pooled  = (float*)carve((size_t)NG * 512 * 4);
    u16* pooledbf  = (u16*)carve((size_t)NG * 512 * 2);

    // ---- weights -> bf16 transposed [out][in]
    k_transpose_bf16<<<ceil_div((long)NF*D_G1,256),256,0,stream>>>(W_g1,  wt_g1,  NF,   D_G1);
    k_transpose_bf16<<<ceil_div((long)D_G1*D_G2,256),256,0,stream>>>(W_g2, wt_g2,  D_G1, D_G2);
    k_transpose_bf16<<<ceil_div((long)D_G2*D_G21,256),256,0,stream>>>(W_g21,wt_g21, D_G2, D_G21);
    k_transpose_bf16<<<ceil_div((long)D_G21*D_E1,256),256,0,stream>>>(W_g32,wt_g32, D_G21,D_E1);
    k_transpose_bf16<<<ceil_div((long)2*NF*D_E1,256),256,0,stream>>>(W_e1, wt_e1,  2*NF, D_E1);
    k_transpose_bf16<<<ceil_div((long)D_E1*D_F1,256),256,0,stream>>>(W_e2, wt_e2,  D_E1, D_F1);
    k_transpose_bf16<<<ceil_div((long)(D_E1+D_F1)*D_F1,256),256,0,stream>>>(W_fc1,wt_fc1,D_E1+D_F1,D_F1);

    // ---- CSR by dst (+ degree norm): cnt -> dinv, rowptr, eidx
    k_fill_i32<<<ceil_div(NN,256),256,0,stream>>>(cnt, NN, 0);
    k_hist<<<ceil_div(NE,256),256,0,stream>>>(dst, cnt, NE);
    k_dinv_from_cnt<<<ceil_div(NN,256),256,0,stream>>>(cnt, dinv, NN);
    k_scan_rowptr<<<1,256,0,stream>>>(cnt, rowptr, NN);
    k_copy_i32<<<ceil_div(NN,256),256,0,stream>>>(rowptr, cursor, NN);
    k_fill_eidx<<<ceil_div(NE,256),256,0,stream>>>(dst, cursor, eidx, NE);

    // ---- initial features to bf16
    k_cvt_bf16<<<ceil_div((long)NN*NF,256),256,0,stream>>>(x, xbf, (long)NN*NF);

    // ---- GCN layers: linear(WMMA) -> fused CSR aggregate + bias + act
    auto run_gcn = [&](const u16* wt, const float* bias, int ind, int outd, int act) {
        dim3 g(outd / 16, ceil_div(NN, 128));
        k_gemm_wmma<<<g, 256, 0, stream>>>(xbf, wt, nullptr, hbuf, NN, ind, outd, 0);
        k_aggregate_act<<<ceil_div(NN,8), 256, 0, stream>>>(
            hbuf, dinv, rowptr, eidx, src, bias, actf, xbf, NN, outd, act);
    };
    run_gcn(wt_g1,  b_g1,  NF,    D_G1,  2);   // sigmoid
    run_gcn(wt_g2,  b_g2,  D_G1,  D_G2,  2);   // sigmoid
    run_gcn(wt_g21, b_g21, D_G2,  D_G21, 2);   // sigmoid
    run_gcn(wt_g32, b_g32, D_G21, D_E1,  3);   // selu -> x1 (fp32) lives in actf

    // ---- EdgeConv (fused 2-layer MLP + max-scatter)
    k_fill_f32<<<ceil_div((long)NN*D_F1,256),256,0,stream>>>(x2, (long)NN*D_F1, 0.0f);
    k_edgeconv<<<ceil_div(NE,16),256,0,stream>>>(x, src, dst, wt_e1, b_e1, wt_e2, b_e2, x2);

    // ---- global add pool over cat[x1, x2]
    k_fill_f32<<<ceil_div((long)NG*512,256),256,0,stream>>>(pooled, (long)NG*512, 0.0f);
    k_pool<<<ceil_div((long)NN*512,256),256,0,stream>>>(actf, x2, batch, pooled, (long)NN*512);

    // ---- fc1 + ReLU -> d_out [512, 256]
    k_cvt_bf16<<<ceil_div((long)NG*512,256),256,0,stream>>>(pooled, pooledbf, (long)NG*512);
    dim3 gf(D_F1 / 16, ceil_div(NG, 128));
    k_gemm_wmma<<<gf, 256, 0, stream>>>(pooledbf, wt_fc1, b_fc1, out, NG, D_E1 + D_F1, D_F1, 1);
}
